// GraphAttention_60954175865456
// MI455X (gfx1250) — compile-verified, matrix operand order checked by reading, not run on hardware
//
#include <hip/hip_runtime.h>

// GAT forward, MI455X (gfx1250, wave32, WMMA + async global->LDS).
// N=8192 nodes, F=128 in-features, F'=64 out-features/head, H=4 heads.

#define NN 8192
#define FF 128
#define FD 64
#define HH 4

typedef __attribute__((ext_vector_type(16))) _Float16 v16h;
typedef __attribute__((ext_vector_type(8)))  float    v8f;

union HF16 {
    v16h v;
    _Float16 e[16];
    unsigned u[8];
};

__device__ __forceinline__ float leaky02(float x) { return x > 0.f ? x : 0.2f * x; }

// ---- CDNA5 async global->LDS (ASYNCcnt) support, with safe fallback ----
#if defined(__gfx1250__) && \
    __has_builtin(__builtin_amdgcn_global_load_async_to_lds_b128) && \
    __has_builtin(__builtin_amdgcn_global_load_async_to_lds_b32) && \
    __has_builtin(__builtin_amdgcn_s_wait_asynccnt)
#define USE_ASYNC 1
// exact parameter types per hipcc diagnostic:
//   b128: (int __vector(4) addrspace(1)*, int __vector(4) addrspace(3)*, imm, imm)
//   b32 : (int addrspace(1)*,             int addrspace(3)*,             imm, imm)
typedef int vi4 __attribute__((vector_size(16)));
typedef __attribute__((address_space(1))) vi4 gvi4;
typedef __attribute__((address_space(3))) vi4 lvi4;
typedef __attribute__((address_space(1))) int gi32;
typedef __attribute__((address_space(3))) int li32;
__device__ __forceinline__ void async_ld_b128(const void* g, void* l) {
    __builtin_amdgcn_global_load_async_to_lds_b128((gvi4*)g, (lvi4*)l, 0, 0);
}
__device__ __forceinline__ void async_ld_b32(const void* g, void* l) {
    __builtin_amdgcn_global_load_async_to_lds_b32((gi32*)g, (li32*)l, 0, 0);
}
#else
#define USE_ASYNC 0
#endif

// ---------------------------------------------------------------------------
// Kernel 1: Xp[h] = X * W[h]  (16x16x32 f16 WMMA), plus
//   s_self[h][n]  = Xp[h][n,:] . a_self[h]
//   s_neigh[h][n] = Xp[h][n,:] . a_neigh[h]
// Xp stored TRANSPOSED as XpT[h][d][n] in fp16 (d-major) so the attention
// kernel's B fragments are contiguous 32-bit words in LDS.
// One wave per (16-row tile, head).
// ---------------------------------------------------------------------------
__global__ void __launch_bounds__(32) k_proj(
    const float* __restrict__ X, const float* __restrict__ W,
    const float* __restrict__ a_self, const float* __restrict__ a_neigh,
    _Float16* __restrict__ XpT, float* __restrict__ s_self,
    float* __restrict__ s_neigh)
{
    const int ln  = threadIdx.x;
    const int h   = blockIdx.y;
    const int n0  = blockIdx.x * 16;
    const int r16 = ln & 15;
    const int kb  = (ln < 16) ? 0 : 8;    // A/B fragment K base (16-bit layout)
    const int row = n0 + r16;

    v8f c[4];
    const v8f vzero = {0.f,0.f,0.f,0.f,0.f,0.f,0.f,0.f};
    #pragma unroll
    for (int nt = 0; nt < 4; ++nt) c[nt] = vzero;

    #pragma unroll
    for (int kk = 0; kk < FF; kk += 32) {
        // A fragment: lane holds row (n0 + ln%16), K = kb+{0..7} and 16+kb+{0..7}
        HF16 a;
        const float* xr = X + (size_t)row * FF + kk + kb;
        float4 x0 = *(const float4*)(xr);
        float4 x1 = *(const float4*)(xr + 4);
        float4 x2 = *(const float4*)(xr + 16);
        float4 x3 = *(const float4*)(xr + 20);
        a.e[0]=(_Float16)x0.x;  a.e[1]=(_Float16)x0.y;  a.e[2]=(_Float16)x0.z;  a.e[3]=(_Float16)x0.w;
        a.e[4]=(_Float16)x1.x;  a.e[5]=(_Float16)x1.y;  a.e[6]=(_Float16)x1.z;  a.e[7]=(_Float16)x1.w;
        a.e[8]=(_Float16)x2.x;  a.e[9]=(_Float16)x2.y;  a.e[10]=(_Float16)x2.z; a.e[11]=(_Float16)x2.w;
        a.e[12]=(_Float16)x3.x; a.e[13]=(_Float16)x3.y; a.e[14]=(_Float16)x3.z; a.e[15]=(_Float16)x3.w;

        #pragma unroll
        for (int nt = 0; nt < 4; ++nt) {
            const int d = nt * 16 + r16;  // B fragment: lane holds column d
            HF16 b;
            #pragma unroll
            for (int i = 0; i < 16; ++i) {
                int k = kk + kb + (i & 7) + ((i >> 3) << 4);
                b.e[i] = (_Float16)W[(size_t)h * FF * FD + (size_t)k * FD + d];
            }
            c[nt] = __builtin_amdgcn_wmma_f32_16x16x32_f16(
                false, a.v, false, b.v, (short)0, c[nt], false, false);
        }
    }

    // ---- attention score vectors (dot with a_self / a_neigh) ----
    float asv[4], anv[4];
    #pragma unroll
    for (int nt = 0; nt < 4; ++nt) {
        asv[nt] = a_self [h * FD + nt * 16 + r16];
        anv[nt] = a_neigh[h * FD + nt * 16 + r16];
    }
    float ps[8], pn[8];
    #pragma unroll
    for (int cc = 0; cc < 8; ++cc) {
        float s = 0.f, n = 0.f;
        #pragma unroll
        for (int nt = 0; nt < 4; ++nt) {
            float v = c[nt][cc];
            s += v * asv[nt];
            n += v * anv[nt];
        }
        ps[cc] = s; pn[cc] = n;
    }
    // reduce across the 16 lanes of each half (rows 0..7 in lanes 0-15, 8..15 in 16-31)
    #pragma unroll
    for (int off = 1; off < 16; off <<= 1) {
        #pragma unroll
        for (int cc = 0; cc < 8; ++cc) {
            ps[cc] += __shfl_xor(ps[cc], off, 32);
            pn[cc] += __shfl_xor(pn[cc], off, 32);
        }
    }
    if (r16 == 0) {  // lanes 0 and 16
        int base = (ln < 16) ? 0 : 8;
        #pragma unroll
        for (int cc = 0; cc < 8; ++cc) {
            s_self [h * NN + n0 + base + cc] = ps[cc];
            s_neigh[h * NN + n0 + base + cc] = pn[cc];
        }
    }

    // ---- store Xp transposed as fp16: XpT[(h*64+d)*N + n] ----
    #pragma unroll
    for (int nt = 0; nt < 4; ++nt) {
        const int d = nt * 16 + r16;
        #pragma unroll
        for (int cc = 0; cc < 8; ++cc) {
            int n = n0 + cc + ((ln < 16) ? 0 : 8);
            XpT[((size_t)(h * FD + d)) * NN + n] = (_Float16)c[nt][cc];
        }
    }
}

// ---------------------------------------------------------------------------
// Kernel 2: per-head max of s_neigh (used as a safe softmax upper bound)
// ---------------------------------------------------------------------------
__global__ void __launch_bounds__(256) k_smax(
    const float* __restrict__ s_neigh, float* __restrict__ smax)
{
    __shared__ float red[256];
    const int h = blockIdx.x, tid = threadIdx.x;
    float m = -3.4e38f;
    for (int i = tid; i < NN; i += 256) m = fmaxf(m, s_neigh[h * NN + i]);
    red[tid] = m;
    __syncthreads();
    for (int s = 128; s > 0; s >>= 1) {
        if (tid < s) red[tid] = fmaxf(red[tid], red[tid + s]);
        __syncthreads();
    }
    if (tid == 0) smax[h] = red[0];
}

// ---------------------------------------------------------------------------
// Kernel 3: streaming masked softmax + P*V via WMMA, double-buffered with
// CDNA5 async global->LDS loads (ASYNCcnt) so tile t+1 streams from HBM
// while tile t runs exp (TRANS pipe) + v_wmma.
// Block = 128 threads = 4 waves; wave w handles head w for one 16-row tile,
// so the adjacency tile is staged ONCE and shared by all 4 heads -> A is
// read from HBM exactly once (268 MB = the bandwidth floor).
// Safe row max M_n = leaky(s_self[n] + max_m s_neigh[m]) (leaky monotone)
// eliminates online rescaling: single pass, exp in [0,1] -> fp16 P frags.
// ---------------------------------------------------------------------------
__global__ void __launch_bounds__(128) k_attn(
    const float* __restrict__ A, const unsigned* __restrict__ XpT_u32,
    const float* __restrict__ s_self, const float* __restrict__ s_neigh,
    const float* __restrict__ smax, float* __restrict__ out)
{
    // padded strides: 68 (f32) and 36 (u32) => rows 16B-aligned (b128-safe)
    // AND bank-conflict-free column reads on the 64-bank LDS.
    __shared__ __align__(16) float    bufA [2][16 * 68];       // 2 x 4.25 KB
    __shared__ __align__(16) float    bufsn[2][HH * 64];       // 2 x 1    KB
    __shared__ __align__(16) unsigned bufxp[2][HH * 64 * 36];  // 2 x 36   KB

    const int tid = threadIdx.x;
    const int ln  = tid & 31;
    const int h   = tid >> 5;              // wave id == head id
    const int n0  = blockIdx.x * 16;
    const int r16 = ln & 15;
    const int kb  = (ln < 16) ? 0 : 8;     // fragment K base
    const int qb  = (ln < 16) ? 0 : 4;     // packed-pair base for B frags

    const float ssl = s_self[h * NN + n0 + r16];
    const float Mn  = leaky02(ssl + smax[h]);   // safe per-row softmax max

    v8f c[4];
    const v8f vzero = {0.f,0.f,0.f,0.f,0.f,0.f,0.f,0.f};
    #pragma unroll
    for (int nt = 0; nt < 4; ++nt) c[nt] = vzero;
    float rowsum = 0.f;

    // stage one 64-column tile (20 async ops per thread => 20 per wave)
    auto stage = [&](int b, int m0) {
        // adjacency tile [16 x 64] f32 (shared by all heads)
        #pragma unroll
        for (int i = 0; i < 2; ++i) {
            int idx = tid + 128 * i;
            int r = idx >> 4, f4 = idx & 15;
            const float* g = A + (size_t)(n0 + r) * NN + m0 + f4 * 4;
            float* dst = &bufA[b][r * 68 + f4 * 4];
#if USE_ASYNC
            async_ld_b128(g, dst);
#else
            *(float4*)dst = *(const float4*)g;
#endif
        }
        // s_neigh tiles [4 heads x 64]
        #pragma unroll
        for (int i = 0; i < 2; ++i) {
            int idx = tid + 128 * i;
            int hh = idx >> 6, j = idx & 63;
            const float* g = s_neigh + hh * NN + m0 + j;
#if USE_ASYNC
            async_ld_b32(g, &bufsn[b][idx]);
#else
            bufsn[b][idx] = *g;
#endif
        }
        // this wave's head Xp tile: 64 d-rows x 32 u32 (64 fp16 m-values)
        #pragma unroll
        for (int i = 0; i < 16; ++i) {
            int idx = ln + 32 * i;
            int d = idx >> 3, f4 = idx & 7;
            const unsigned* g = XpT_u32 +
                ((size_t)(h * FD + d)) * (NN / 2) + (m0 >> 1) + f4 * 4;
            unsigned* dst = &bufxp[b][(h * FD + d) * 36 + f4 * 4];
#if USE_ASYNC
            async_ld_b128(g, dst);
#else
            *(uint4*)dst = *(const uint4*)g;
#endif
        }
    };

    stage(0, 0);  // prologue: tile 0 in flight

    #pragma unroll 1
    for (int t = 0; t < NN / 64; ++t) {
        const int p = t & 1;
        const bool hasnext = (t + 1) < (NN / 64);
        if (hasnext) stage(p ^ 1, (t + 1) * 64);   // overlap: tile t+1 streams
#if USE_ASYNC
        if (hasnext) __builtin_amdgcn_s_wait_asynccnt(20);  // tile t landed
        else         __builtin_amdgcn_s_wait_asynccnt(0);
#endif
        __syncthreads();   // all waves see tile t in LDS

        #pragma unroll
        for (int kk2 = 0; kk2 < 2; ++kk2) {
            const int kbase = kk2 * 32;
            // ---- P fragment: p = A * exp(leaky(s_self+s_neigh) - Mn) ----
            HF16 a;
            #pragma unroll
            for (int i = 0; i < 16; ++i) {
                int j = kbase + kb + (i & 7) + ((i >> 3) << 4);
                float s = leaky02(ssl + bufsn[p][h * 64 + j]);
                float pv = __expf(s - Mn) * bufA[p][r16 * 68 + j]; // A in {0,1}
                rowsum += pv;
                a.e[i] = (_Float16)pv;
            }
            // ---- 4 WMMAs: C[16x64] += P[16x32] * Xp[32x64] ----
            #pragma unroll
            for (int nt = 0; nt < 4; ++nt) {
                HF16 b;
                #pragma unroll
                for (int jj = 0; jj < 8; ++jj) {
                    int q = (kbase >> 1) + qb + (jj & 3) + ((jj >> 2) << 3);
                    b.u[jj] = bufxp[p][(h * FD + nt * 16 + r16) * 36 + q];
                }
                c[nt] = __builtin_amdgcn_wmma_f32_16x16x32_f16(
                    false, a.v, false, b.v, (short)0, c[nt], false, false);
            }
        }
        __syncthreads();   // buffer p free for restaging next iteration
    }

    // ---- normalize, ReLU, scatter to [N, H*F'] ----
    float full = rowsum + __shfl_xor(rowsum, 16, 32);  // lane: sum of row ln%16
    #pragma unroll
    for (int cc = 0; cc < 8; ++cc) {
        int rowm = cc + ((ln < 16) ? 0 : 8);
        float rs  = __shfl(full, rowm, 32);
        float inv = 1.f / rs;
        int n = n0 + rowm;
        #pragma unroll
        for (int nt = 0; nt < 4; ++nt) {
            float v = c[nt][cc] * inv;
            out[(size_t)n * (HH * FD) + h * FD + nt * 16 + r16] = v > 0.f ? v : 0.f;
        }
    }
}

// ---------------------------------------------------------------------------
extern "C" void kernel_launch(void* const* d_in, const int* in_sizes, int n_in,
                              void* d_out, int out_size, void* d_ws, size_t ws_size,
                              hipStream_t stream)
{
    const float* X       = (const float*)d_in[0];
    const float* A       = (const float*)d_in[1];
    const float* W       = (const float*)d_in[2];
    const float* a_self  = (const float*)d_in[3];
    const float* a_neigh = (const float*)d_in[4];
    float* out = (float*)d_out;

    // workspace layout (~4.46 MB total)
    char* ws = (char*)d_ws;
    _Float16* XpT   = (_Float16*)ws;                          // H*64*N fp16 = 4 MB
    float* s_self   = (float*)(ws + (size_t)HH * FD * NN * 2);
    float* s_neigh  = s_self + (size_t)HH * NN;
    float* smax     = s_neigh + (size_t)HH * NN;

    k_proj<<<dim3(NN / 16, HH), 32, 0, stream>>>(X, W, a_self, a_neigh,
                                                 XpT, s_self, s_neigh);
    k_smax<<<HH, 256, 0, stream>>>(s_neigh, smax);
    k_attn<<<NN / 16, 128, 0, stream>>>(A, (const unsigned*)XpT,
                                        s_self, s_neigh, smax, out);
}
